// LearnedSimulator_64467459113218
// MI455X (gfx1250) — matrix operand order repaired and checked by reading, not compiled, not run on hardware
//
#include <hip/hip_runtime.h>
#include <hip/hip_bf16.h>

typedef __attribute__((ext_vector_type(16))) _Float16 v16h;
typedef __attribute__((ext_vector_type(8)))  float    v8f;

#define TPB   128
#define NWAVE 4

// ---------------------------------------------------------------------------
// Generic fused 3-layer MLP (in -> 64 -> 64 -> OUTN), relu between layers,
// optional LayerNorm, optional antisymmetrization 0.5*(f(x)-f(-x)).
// One wave computes a 16-row tile; all GEMMs via v_wmma_f32_16x16x32_f16.
// Layer-1 weights staged to LDS with GLOBAL_LOAD_ASYNC_TO_LDS_B128.
// ---------------------------------------------------------------------------
struct MlpArgs {
  const float *W0, *W1, *W2;      // (k_in,64) (64,64) (64,OUTN) row-major
  const float *b0, *b1, *b2;
  const float *g, *beta;
  const float *in_a, *in_b, *embed;
  const int   *src, *dst, *ptype;
  float *out_ptr;
  float *aggr;
  int rows, k_in;
};

__device__ __forceinline__ int frag_kbase(int v, int half) {
  // 16-bit A-matrix 16x32 layout (ISA 7.12.2): VGPR v halves -> K offsets
  return ((v < 4) ? (2 * v) : (16 + 2 * (v - 4))) + half * 8;
}

__device__ __forceinline__ v8f wmma_f16(v16h a, v16h b, v8f c) {
  return __builtin_amdgcn_wmma_f32_16x16x32_f16(false, a, false, b, (short)0, c,
                                                false, false);
}

// B fragment (32x16 f16): lane%16 = N within tile, VGPR halves = K (same order as A)
__device__ __forceinline__ v16h bfrag_lds16(const _Float16* W, int lane, int kt, int nt) {
  v16h b;
  const _Float16* p = W + (kt * 32) * 64 + nt * 16 + (lane & 15);
#pragma unroll
  for (int v = 0; v < 8; ++v) { int k = frag_kbase(v, lane >> 4); b[2 * v] = p[k * 64]; b[2 * v + 1] = p[(k + 1) * 64]; }
  return b;
}

__device__ __forceinline__ v16h bfrag_lds32(const float* W, int lane, int kt, int nt) {
  v16h b;
  const float* p = W + (kt * 32) * 64 + nt * 16 + (lane & 15);
#pragma unroll
  for (int v = 0; v < 8; ++v) {
    int k = frag_kbase(v, lane >> 4);
    b[2 * v]     = (_Float16)p[k * 64];
    b[2 * v + 1] = (_Float16)p[(k + 1) * 64];
  }
  return b;
}

template<int OUTN>
__device__ __forceinline__ v16h bfrag_glb(const float* W, int lane, int kt, int nt) {
  v16h b;
  const float* p = W + (kt * 32) * OUTN;
  int c0 = nt * 16 + (lane & 15);
#pragma unroll
  for (int v = 0; v < 8; ++v) {
    int k = frag_kbase(v, lane >> 4);
    b[2 * v]     = (c0 < OUTN) ? (_Float16)p[k * OUTN + c0]       : (_Float16)0.f;
    b[2 * v + 1] = (c0 < OUTN) ? (_Float16)p[(k + 1) * OUTN + c0] : (_Float16)0.f;
  }
  return b;
}

// A fragment (16x32 f16) from a row-major [16][64] f16 LDS tile
__device__ __forceinline__ v16h afrag_lds(const _Float16* Hb, int lane, int kt) {
  v16h a;
  int m = lane & 15, half = lane >> 4;
#pragma unroll
  for (int v = 0; v < 8; ++v) {
    int k = kt * 32 + frag_kbase(v, half);
    a[2 * v]     = Hb[m * 64 + k];
    a[2 * v + 1] = Hb[m * 64 + k + 1];
  }
  return a;
}

template<int IM>
__device__ __forceinline__ float in_val(const MlpArgs& A, int ia, int ib, int ic, int k) {
  if (k >= A.k_in) return 0.0f;
  if constexpr (IM == 0) {
    return A.in_a[ia + k];
  } else if constexpr (IM == 1) {          // concat(xd-xs, xs-xd, ef)
    if (k < 64)  return A.in_a[ia + k] - A.in_a[ib + k];
    if (k < 128) return A.in_a[ib + (k - 64)] - A.in_a[ia + (k - 64)];
    return A.in_b[ic + (k - 128)];
  } else if constexpr (IM == 2) {          // concat(xd, xs, ef)
    if (k < 64)  return A.in_a[ia + k];
    if (k < 128) return A.in_a[ib + (k - 64)];
    return A.in_b[ic + (k - 128)];
  } else if constexpr (IM == 3) {          // concat(x, aggr)
    return (k < 64) ? A.in_a[ia + k] : A.in_b[ia + (k - 64)];
  } else {                                 // concat(embed[type], out2)
    return (k < 16) ? A.embed[ia + k] : A.in_b[ib + (k - 16)];
  }
}

template<bool NEG, int NKT, int LNON, int OUTN>
__device__ __forceinline__ void mlp_pass(const MlpArgs& A, const v16h* afr,
                                         const _Float16* sW0, const float* sW1,
                                         _Float16* sHw, int lane, float (*res)[8]) {
  constexpr int NT2 = (OUTN + 15) / 16;
  const int m = lane & 15, half = lane >> 4;
  const v8f zf = {};
  // ---- layer 0: (16 x 32*NKT) @ W0, bias + relu -> sHw
  v8f acc[4] = {zf, zf, zf, zf};
#pragma unroll
  for (int t = 0; t < NKT; ++t) {
    v16h a = afr[t];
    if constexpr (NEG) {
#pragma unroll
      for (int i = 0; i < 16; ++i) a[i] = -a[i];
    }
#pragma unroll
    for (int nt = 0; nt < 4; ++nt)
      acc[nt] = wmma_f16(a, bfrag_lds16(sW0, lane, t, nt), acc[nt]);
  }
#pragma unroll
  for (int nt = 0; nt < 4; ++nt) {
    float bb = A.b0[nt * 16 + m];
#pragma unroll
    for (int v = 0; v < 8; ++v)
      sHw[(v + 8 * half) * 64 + nt * 16 + m] = (_Float16)fmaxf(acc[nt][v] + bb, 0.f);
  }
  // ---- layer 1: 64 @ W1 (f32 LDS, async-staged), bias + relu -> sHw
#pragma unroll
  for (int nt = 0; nt < 4; ++nt) acc[nt] = zf;
#pragma unroll
  for (int t = 0; t < 2; ++t) {
    v16h a = afrag_lds(sHw, lane, t);
#pragma unroll
    for (int nt = 0; nt < 4; ++nt)
      acc[nt] = wmma_f16(a, bfrag_lds32(sW1, lane, t, nt), acc[nt]);
  }
#pragma unroll
  for (int nt = 0; nt < 4; ++nt) {
    float bb = A.b1[nt * 16 + m];
#pragma unroll
    for (int v = 0; v < 8; ++v)
      sHw[(v + 8 * half) * 64 + nt * 16 + m] = (_Float16)fmaxf(acc[nt][v] + bb, 0.f);
  }
  // ---- layer 2: 64 @ W2 (global), bias, optional LN
  v8f acc2[NT2];
#pragma unroll
  for (int nt = 0; nt < NT2; ++nt) acc2[nt] = zf;
#pragma unroll
  for (int t = 0; t < 2; ++t) {
    v16h a = afrag_lds(sHw, lane, t);
#pragma unroll
    for (int nt = 0; nt < NT2; ++nt)
      acc2[nt] = wmma_f16(a, bfrag_glb<OUTN>(A.W2, lane, t, nt), acc2[nt]);
  }
  float fin[NT2][8];
#pragma unroll
  for (int nt = 0; nt < NT2; ++nt) {
    int cn = nt * 16 + m;
    float bb = (cn < OUTN) ? A.b2[cn] : 0.f;
#pragma unroll
    for (int v = 0; v < 8; ++v) fin[nt][v] = acc2[nt][v] + bb;
  }
  if constexpr (LNON) {   // LayerNorm over 64 cols: row M lives at one VGPR idx
                          // across a 16-lane group -> shfl_xor tree reduction
#pragma unroll
    for (int v = 0; v < 8; ++v) {
      float s = 0.f, s2 = 0.f;
#pragma unroll
      for (int nt = 0; nt < 4; ++nt) { float x = fin[nt][v]; s += x; s2 += x * x; }
#pragma unroll
      for (int w = 1; w < 16; w <<= 1) {
        s  += __shfl_xor(s,  w, 32);
        s2 += __shfl_xor(s2, w, 32);
      }
      float mu = s * (1.f / 64.f);
      float rs = rsqrtf(s2 * (1.f / 64.f) - mu * mu + 1e-5f);
#pragma unroll
      for (int nt = 0; nt < 4; ++nt) {
        int cn = nt * 16 + m;
        fin[nt][v] = (fin[nt][v] - mu) * rs * A.g[cn] + A.beta[cn];
      }
    }
  }
#pragma unroll
  for (int nt = 0; nt < NT2; ++nt)
#pragma unroll
    for (int v = 0; v < 8; ++v) res[nt][v] = fin[nt][v];
}

template<int IM, int OM, int AN, int LNON, int NKT, int OUTN>
__global__ void __launch_bounds__(TPB) mlp3_kernel(MlpArgs A) {
  constexpr int NT2 = (OUTN + 15) / 16;
  __shared__ _Float16 sW0[NKT * 32 * 64];                       // <= 24 KB (f16)
  __shared__ __align__(16) float sW1[64 * 64];                  // 16 KB (f32, async)
  __shared__ _Float16 sH[NWAVE][16 * 64];                       // 8 KB

  const int tid = threadIdx.x, wave = tid >> 5, lane = tid & 31;
  const int m = lane & 15, half = lane >> 4;
  const int row0 = blockIdx.x * (NWAVE * 16) + wave * 16;

  // ---- kick off async global->LDS copy of W1 (tracked by ASYNCcnt)
  {
    unsigned lbase = (unsigned)(uintptr_t)(void*)&sW1[0];
    for (int i = tid; i < 1024; i += TPB) {          // 1024 x 16B
      unsigned laddr = lbase + (unsigned)i * 16u;
      unsigned long long ga = (unsigned long long)(uintptr_t)(A.W1 + i * 4);
      asm volatile("global_load_async_to_lds_b128 %0, %1, off"
                   :: "v"(laddr), "v"(ga) : "memory");
    }
  }
  // ---- prefetch W2 (global_prefetch_b8)
  if constexpr (OUTN == 64) {
    __builtin_prefetch(A.W2 + tid * 32, 0, 1);
  } else {
    if (tid < 4) __builtin_prefetch(A.W2 + tid * 32, 0, 1);
  }
  // ---- stage W0 into LDS as f16 (zero-pad K)
  const int kin = A.k_in;
  for (int i = tid; i < NKT * 32 * 64; i += TPB) {
    int r = i >> 6, c = i & 63;
    sW0[i] = (r < kin) ? (_Float16)A.W0[r * 64 + c] : (_Float16)0.f;
  }
  __syncthreads();

  // ---- per-lane row context + layer-0 A fragments (row = row0 + m)
  const int r = row0 + m;
  const bool rv = (r < A.rows);
  int ia = 0, ib = 0, ic = 0;
  if (rv) {
    if constexpr (IM == 1 || IM == 2) { ia = A.dst[r] * 64; ib = A.src[r] * 64; ic = r * 64; }
    else if constexpr (IM == 3)       { ia = r * 64; }
    else if constexpr (IM == 4)       { ia = A.ptype[r] * 16; ib = r * 2; }
    else                              { ia = r * A.k_in; }
  }
  v16h afr[NKT];
#pragma unroll
  for (int t = 0; t < NKT; ++t) {
    v16h a;
#pragma unroll
    for (int v = 0; v < 8; ++v) {
      int k = t * 32 + frag_kbase(v, half);
      float x0 = rv ? in_val<IM>(A, ia, ib, ic, k) : 0.f;
      float x1 = rv ? in_val<IM>(A, ia, ib, ic, k + 1) : 0.f;
      a[2 * v]     = (_Float16)x0;
      a[2 * v + 1] = (_Float16)x1;
    }
    afr[t] = a;
  }

  // ---- wait for async W1 staging, make visible to all waves
  asm volatile("s_wait_asynccnt 0x0" ::: "memory");
  __syncthreads();

  // ---- run MLP (x), and for antisym also (-x); combine 0.5*(f(x)-f(-x))
  float res[NT2][8];
  mlp_pass<false, NKT, LNON, OUTN>(A, afr, sW0, sW1, sH[wave], lane, res);
  if constexpr (AN) {
    float res2[NT2][8];
    mlp_pass<true, NKT, LNON, OUTN>(A, afr, sW0, sW1, sH[wave], lane, res2);
#pragma unroll
    for (int nt = 0; nt < NT2; ++nt)
#pragma unroll
      for (int v = 0; v < 8; ++v) res[nt][v] = 0.5f * (res[nt][v] - res2[nt][v]);
  }

  // ---- emit (C layout: lane -> N = lane&15, M = v + 8*(lane>>4))
#pragma unroll
  for (int nt = 0; nt < NT2; ++nt) {
    int cn = nt * 16 + m;
#pragma unroll
    for (int v = 0; v < 8; ++v) {
      int cm = v + 8 * half;
      int rr = row0 + cm;
      if (rr < A.rows && cn < OUTN) {
        float val = res[nt][v];
        if constexpr (OM == 0) {
          A.out_ptr[rr * OUTN + cn] = val;
        } else if constexpr (OM == 1) {    // ef += msg ; scatter msg to aggr[dst]
          A.out_ptr[rr * 64 + cn] = A.in_b[rr * 64 + cn] + val;
          atomicAdd(&A.aggr[A.dst[rr] * 64 + cn], val);
        } else {                           // residual add
          A.out_ptr[rr * OUTN + cn] += val;
        }
      }
    }
  }
}

// ---------------------------------------------------------------------------
// Small elementwise / scatter kernels
// ---------------------------------------------------------------------------
__global__ void zero_kernel(float* p, int n) {
  int i = blockIdx.x * blockDim.x + threadIdx.x;
  if (i < n) p[i] = 0.f;
}

__global__ void stage1_bias_kernel(float* out, const float* down, const float* accs, int n2) {
  int i = blockIdx.x * blockDim.x + threadIdx.x;
  if (i >= n2) return;
  float g = 5.5339e-05f / accs[0];
  out[i] += g * down[i & 1];
}

__global__ void obstacle_scatter_kernel(const float* approach, const float* invd,
                                        const float* dir, const int* tgt,
                                        const float* accs, const float* vels,
                                        float* ob, int ne2) {
  int i = blockIdx.x * blockDim.x + threadIdx.x;
  if (i >= ne2) return;
  int e = i >> 1, d = i & 1;
  float vscale = vels[0] / accs[0] * (60.0f / 94.0f);
  float val = invd[e] * approach[e] * dir[e * 2 + d] * (vscale * 10.0f);
  atomicAdd(&ob[tgt[e] * 2 + d], val);
}

__global__ void node_bias_kernel(float* out, const float* ob, const float* rv,
                                 const float* nid, const float* nds,
                                 const float* ux, const float* uy,
                                 const float* accs, const float* vels, int n) {
  int i = blockIdx.x * blockDim.x + threadIdx.x;
  if (i >= n) return;
  float vscale = vels[0] / accs[0] * (60.0f / 94.0f);
  float o0 = ob[i * 2], o1 = ob[i * 2 + 1];
  float c0 = fabsf(rv[i * 2] * vscale), c1 = fabsf(rv[i * 2 + 1] * vscale);
  o0 = (o0 > 0.f ? 1.f : (o0 < 0.f ? -1.f : 0.f)) * fminf(fabsf(o0), c0);
  o1 = (o1 > 0.f ? 1.f : (o1 < 0.f ? -1.f : 0.f)) * fminf(fabsf(o1), c1);
  float w0 = 0.f, w1 = 0.f;
  float vv[4] = { fminf(fmaxf(-rv[i * 2], 0.f), 1000.f),
                  fminf(fmaxf(-rv[i * 2 + 1], 0.f), 1000.f),
                  fminf(fmaxf( rv[i * 2], 0.f), 1000.f),
                  fminf(fmaxf( rv[i * 2 + 1], 0.f), 1000.f) };
  float dx[4] = { -ux[0], -uy[0], ux[0], uy[0] };
  float dy[4] = { -ux[1], -uy[1], ux[1], uy[1] };
  for (int k = 0; k < 4; ++k) {
    float mask = (nid[i * 4 + k] > 1e-7f) ? 1.f : 0.f;
    float factor = (nds[i * 4 + k] < 0.f) ? 1.f : nid[i * 4 + k];
    float c = mask * factor * vv[k] * vscale;
    w0 += c * dx[k];
    w1 += c * dy[k];
  }
  out[i * 2]     -= o0 + w0;
  out[i * 2 + 1] -= o1 + w1;
}

// ---------------------------------------------------------------------------
// Host orchestration
// ---------------------------------------------------------------------------
struct MlpW { const float *W0, *W1, *W2, *b0, *b1, *b2, *beta, *g; };

template<int IM, int OM, int AN, int LNON, int NKT, int OUTN>
static inline void run_mlp(const MlpW& w, const float* ina, const float* inb,
                           const int* src, const int* dst, const int* ptype,
                           const float* embed, int rows, int k_in,
                           float* out_ptr, float* aggr, hipStream_t stream) {
  MlpArgs A;
  A.W0 = w.W0; A.W1 = w.W1; A.W2 = w.W2;
  A.b0 = w.b0; A.b1 = w.b1; A.b2 = w.b2;
  A.g = w.g; A.beta = w.beta;
  A.in_a = ina; A.in_b = inb; A.embed = embed;
  A.src = src; A.dst = dst; A.ptype = ptype;
  A.out_ptr = out_ptr; A.aggr = aggr;
  A.rows = rows; A.k_in = k_in;
  mlp3_kernel<IM, OM, AN, LNON, NKT, OUTN><<<(rows + 63) / 64, TPB, 0, stream>>>(A);
}

extern "C" void kernel_launch(void* const* d_in, const int* in_sizes, int n_in,
                              void* d_out, int out_size, void* d_ws, size_t ws_size,
                              hipStream_t stream) {
  (void)in_sizes; (void)out_size; (void)ws_size;
  const int N = 10000, E = 100000, E2 = 100000, E3 = 20000, NMP = 10;
  if (n_in < 374) return;   // parameter-pytree layout assumption not met

  const float* edge_attr  = (const float*)d_in[0];
  const float* edge_attr2 = (const float*)d_in[1];
  const float* approach   = (const float*)d_in[2];
  const float* invd3      = (const float*)d_in[3];
  const float* dir3       = (const float*)d_in[4];
  const float* rvel       = (const float*)d_in[5];
  const float* nid        = (const float*)d_in[6];
  const float* nds        = (const float*)d_in[7];
  const float* down       = (const float*)d_in[8];
  const float* unitx      = (const float*)d_in[9];
  const float* unity      = (const float*)d_in[10];
  const float* accs       = (const float*)d_in[11];
  const float* vels       = (const float*)d_in[12];

  int pi = 13;
  auto take = [&](bool ln) -> MlpW {
    MlpW mw;
    mw.W0 = (const float*)d_in[pi];     mw.W1 = (const float*)d_in[pi + 1];
    mw.W2 = (const float*)d_in[pi + 2]; mw.b0 = (const float*)d_in[pi + 3];
    mw.b1 = (const float*)d_in[pi + 4]; mw.b2 = (const float*)d_in[pi + 5];
    if (ln) { mw.beta = (const float*)d_in[pi + 6]; mw.g = (const float*)d_in[pi + 7]; pi += 8; }
    else    { mw.beta = nullptr; mw.g = nullptr; pi += 6; }
    return mw;
  };
  // params pytree, dict keys sorted: edge_in1, edge_in2, embed, layers1, layers2,
  // node_in2, node_out1, node_out2 ; per-MLP leaf order: W0..2, b0..2, beta, g
  MlpW edge_in1 = take(true);
  MlpW edge_in2 = take(true);
  const float* embed = (const float*)d_in[pi]; pi += 1;
  MlpW l1e[NMP], l1n[NMP], l2e[NMP], l2n[NMP];
  for (int i = 0; i < NMP; ++i) { l1e[i] = take(true); l1n[i] = take(true); }
  for (int i = 0; i < NMP; ++i) { l2e[i] = take(true); l2n[i] = take(true); }
  MlpW node_in2  = take(true);
  MlpW node_out1 = take(false);
  MlpW node_out2 = take(false);
  const int* ptype = (const int*)d_in[pi];
  const int* ei1   = (const int*)d_in[pi + 1];
  const int* ei2   = (const int*)d_in[pi + 2];
  const int* ei3   = (const int*)d_in[pi + 3];

  float* ws   = (float*)d_ws;
  float* ef   = ws;                       // E*64
  float* nf   = ef + (size_t)E * 64;      // N*64
  float* aggr = nf + (size_t)N * 64;      // N*64
  float* ob   = aggr + (size_t)N * 64;    // N*2
  float* outp = (float*)d_out;            // N*2

  auto zero = [&](float* p, int n) {
    zero_kernel<<<(n + 255) / 256, 256, 0, stream>>>(p, n);
  };

  // ---------------- stage 1 (antisymmetric) ----------------
  zero(nf, N * 64);
  run_mlp<0, 0, 1, 1, 1, 64>(edge_in1, edge_attr, nullptr, nullptr, nullptr, nullptr,
                             embed, E, 14, ef, nullptr, stream);
  for (int i = 0; i < NMP; ++i) {
    zero(aggr, N * 64);
    run_mlp<1, 1, 1, 1, 6, 64>(l1e[i], nf, ef, ei1, ei1 + E, nullptr,
                               embed, E, 192, ef, aggr, stream);
    run_mlp<3, 2, 1, 1, 4, 64>(l1n[i], nf, aggr, nullptr, nullptr, nullptr,
                               embed, N, 128, nf, nullptr, stream);
  }
  run_mlp<0, 0, 1, 0, 2, 2>(node_out1, nf, nullptr, nullptr, nullptr, nullptr,
                            embed, N, 64, outp, nullptr, stream);
  stage1_bias_kernel<<<(N * 2 + 255) / 256, 256, 0, stream>>>(outp, down, accs, N * 2);
  zero(ob, N * 2);
  obstacle_scatter_kernel<<<(E3 * 2 + 255) / 256, 256, 0, stream>>>(
      approach, invd3, dir3, ei3, accs, vels, ob, E3 * 2);
  node_bias_kernel<<<(N + 255) / 256, 256, 0, stream>>>(
      outp, ob, rvel, nid, nds, unitx, unity, accs, vels, N);

  // ---------------- stage 2 ----------------
  run_mlp<4, 0, 0, 1, 1, 64>(node_in2, nullptr, outp, nullptr, nullptr, ptype,
                             embed, N, 18, nf, nullptr, stream);
  run_mlp<0, 0, 0, 1, 1, 64>(edge_in2, edge_attr2, nullptr, nullptr, nullptr, nullptr,
                             embed, E2, 14, ef, nullptr, stream);
  for (int i = 0; i < NMP; ++i) {
    zero(aggr, N * 64);
    run_mlp<2, 1, 0, 1, 6, 64>(l2e[i], nf, ef, ei2, ei2 + E2, nullptr,
                               embed, E2, 192, ef, aggr, stream);
    run_mlp<3, 2, 0, 1, 4, 64>(l2n[i], nf, aggr, nullptr, nullptr, nullptr,
                               embed, N, 128, nf, nullptr, stream);
  }
  run_mlp<0, 2, 0, 0, 2, 2>(node_out2, nf, nullptr, nullptr, nullptr, nullptr,
                            embed, N, 64, outp, nullptr, stream);
}
